// SplitAttention_74483322847372
// MI455X (gfx1250) — matrix operand order repaired and checked by reading, not compiled, not run on hardware
//
#include <hip/hip_runtime.h>
#include <hip/hip_bf16.h>

// ---------------- problem constants ----------------
#define BATCH 2
#define SEQ   2048
#define DIM   768
#define HEADS 12
#define DHEAD 64
#define MROWS (BATCH * SEQ)      // 4096
#define NQKV  (3 * DIM)          // 2304

typedef __attribute__((ext_vector_type(16))) __bf16 bf16x16;
typedef __attribute__((ext_vector_type(8)))  float  f32x8;

union FragAB {
    unsigned int u[8];
    bf16x16      v;
};

__device__ __forceinline__ f32x8 zero8() {
    f32x8 z = {0.f, 0.f, 0.f, 0.f, 0.f, 0.f, 0.f, 0.f};
    return z;
}

__device__ __forceinline__ unsigned short f2bf(float f) {
    unsigned int u = __float_as_uint(f);
    unsigned int r = u + 0x7FFFu + ((u >> 16) & 1u);   // round-to-nearest-even
    return (unsigned short)(r >> 16);
}

// D = A(16x32 bf16) x B(32x16 bf16) + C(16x16 f32)
__device__ __forceinline__ f32x8 wmma_bf16(const FragAB& a, const FragAB& b, f32x8 c) {
    return __builtin_amdgcn_wmma_f32_16x16x32_bf16(
        /*neg_a=*/false, a.v, /*neg_b=*/false, b.v,
        /*c_mod=*/(short)0, c, /*reuse_a=*/false, /*reuse_b=*/false);
}

// A-fragment K offset for VGPR j, lane-half hi (contiguous k = 16*hi + 2*j, matching B reads)
__device__ __forceinline__ int kOffA(int j, int hi) {
    return hi * 16 + j * 2;
}

// ---------------- packing kernels ----------------
__global__ void pack_hidden_kernel(const float* __restrict__ x,
                                   unsigned short* __restrict__ xb, int n) {
    for (int i = blockIdx.x * blockDim.x + threadIdx.x; i < n; i += gridDim.x * blockDim.x)
        xb[i] = f2bf(x[i]);
}

// WTqkv[n][k] = W_{mat}[k][col]  (n-major, k contiguous) ; WTo[n][k] = Wo[k][n]
__global__ void pack_weights_kernel(const float* __restrict__ Wq, const float* __restrict__ Wk,
                                    const float* __restrict__ Wv, const float* __restrict__ Wo,
                                    unsigned short* __restrict__ WTqkv,
                                    unsigned short* __restrict__ WTo) {
    int i = blockIdx.x * blockDim.x + threadIdx.x;   // over (NQKV + DIM) * DIM
    if (i >= (NQKV + DIM) * DIM) return;
    int n = i / DIM, k = i % DIM;
    if (n < NQKV) {
        int mat = n / DIM, col = n % DIM;
        const float* W = (mat == 0) ? Wq : (mat == 1) ? Wk : Wv;
        WTqkv[(size_t)n * DIM + k] = f2bf(W[(size_t)k * DIM + col]);
    } else {
        int col = n - NQKV;
        WTo[(size_t)col * DIM + k] = f2bf(Wo[(size_t)k * DIM + col]);
    }
}

__global__ void pack_bias_kernel(const float* __restrict__ bq, const float* __restrict__ bk,
                                 const float* __restrict__ bv, float* __restrict__ bqkv) {
    int i = blockIdx.x * blockDim.x + threadIdx.x;
    if (i < DIM)            bqkv[i] = bq[i];
    else if (i < 2 * DIM)   bqkv[i] = bk[i - DIM];
    else if (i < 3 * DIM)   bqkv[i] = bv[i - 2 * DIM];
}

// ---------------- fused QKV projection GEMM ----------------
// Block tile 128(M) x 64(N): 4 waves x (32x64 each). B k-slice staged in LDS,
// shared by all 4 waves; each B fragment feeds 2 WMMAs (2 M sub-tiles).
__global__ void __launch_bounds__(128)
qkv_gemm_kernel(const unsigned short* __restrict__ Xbf,
                const unsigned short* __restrict__ WT,
                const float* __restrict__ bqkv,
                unsigned short* __restrict__ Qh,
                unsigned short* __restrict__ Kh,
                unsigned short* __restrict__ Vh) {
    // 64 rows (n) x 32 k of bf16, padded to 34 shorts (17 dwords) per row: no bank conflicts
    __shared__ __align__(16) unsigned short Bs[64 * 34];

    const int w = threadIdx.x >> 5;
    const int lane = threadIdx.x & 31;
    const int hi = lane >> 4, ln = lane & 15;

    const int bm = blockIdx.x / (NQKV / 64);   // 0..31
    const int bn = blockIdx.x % (NQKV / 64);   // 0..35
    const int m0 = bm * 128 + w * 32;
    const int n0 = bn * 64;

    f32x8 acc[2][4] = {{zero8(), zero8(), zero8(), zero8()},
                       {zero8(), zero8(), zero8(), zero8()}};

    for (int k0 = 0; k0 < DIM; k0 += 32) {
        __syncthreads();
        // stage B k-slice: 64 rows x 16 dwords -> Bs (17-dword padded rows)
        for (int i = threadIdx.x; i < 1024; i += 128) {
            int n = i >> 4, dp = i & 15;
            ((unsigned int*)Bs)[n * 17 + dp] =
                *(const unsigned int*)(WT + (size_t)(n0 + n) * DIM + k0 + dp * 2);
        }
        __syncthreads();

        FragAB a[2];
#pragma unroll
        for (int r = 0; r < 2; r++)
#pragma unroll
            for (int j = 0; j < 8; j++) {
                int kk = k0 + kOffA(j, hi);
                a[r].u[j] = *(const unsigned int*)(Xbf + (size_t)(m0 + r * 16 + ln) * DIM + kk);
            }
#pragma unroll
        for (int dt = 0; dt < 4; dt++) {
            FragAB bfr;
#pragma unroll
            for (int j = 0; j < 8; j++) {
                int kk = hi * 16 + 2 * j;
                bfr.u[j] = *(const unsigned int*)(Bs + (dt * 16 + ln) * 34 + kk);
            }
            acc[0][dt] = wmma_bf16(a[0], bfr, acc[0][dt]);
            acc[1][dt] = wmma_bf16(a[1], bfr, acc[1][dt]);
        }
    }

#pragma unroll
    for (int r = 0; r < 2; r++)
#pragma unroll
        for (int dt = 0; dt < 4; dt++)
#pragma unroll
            for (int j = 0; j < 8; j++) {
                int row = m0 + r * 16 + j + 8 * hi;   // global M row
                int n = n0 + dt * 16 + ln;            // global N col
                float val = acc[r][dt][j] + bqkv[n];
                int mat = n / DIM, col = n % DIM;
                if (mat == 0) val *= 0.125f;          // fold 1/sqrt(Dh) into Q
                int b = row >> 11, s = row & (SEQ - 1);
                int h = col >> 6, d = col & 63;
                size_t idx = (((size_t)(b * HEADS + h) * SEQ) + s) * DHEAD + d;
                unsigned short bv16 = f2bf(val);
                if (mat == 0) Qh[idx] = bv16;
                else if (mat == 1) Kh[idx] = bv16;
                else Vh[idx] = bv16;
            }
}

// ---------------- split-softmax flash-style attention ----------------
// Block = 4 waves, 64 queries of one (b,h). Two passes over K: pass1 builds
// per-row Z_fg/Z_bg, pass2 forms normalized probs and accumulates PV via WMMA.
// LDS rows padded to odd dword strides -> conflict-free fragment reads.
__global__ void __launch_bounds__(128)
attn_kernel(const unsigned short* __restrict__ Qh,
            const unsigned short* __restrict__ Kh,
            const unsigned short* __restrict__ Vh,
            const unsigned char* __restrict__ mask,
            unsigned short* __restrict__ ctxb) {
    __shared__ __align__(16) unsigned short Kc[32 * 66];      // [key][d], 66-short rows
    __shared__ __align__(16) unsigned short Vt[64 * 34];      // [d][key] transposed, 34-short rows
    __shared__ __align__(16) unsigned short Pb[4][16 * 34];   // per-wave P staging

    const int w = threadIdx.x >> 5;
    const int lane = threadIdx.x & 31;
    const int hi = lane >> 4, ln = lane & 15;

    const int bh = blockIdx.x >> 5;       // 0..23  (b*H+h)
    const int qg = blockIdx.x & 31;       // 0..31
    const int b = bh / HEADS, h = bh % HEADS;
    const int q0 = qg * 64 + w * 16;      // this wave's 16 query rows

    const unsigned short* Qbase = Qh + (size_t)bh * SEQ * DHEAD;
    const unsigned short* Kbase = Kh + (size_t)bh * SEQ * DHEAD;
    const unsigned short* Vbase = Vh + (size_t)bh * SEQ * DHEAD;
    const unsigned char*  mrow  = mask + (size_t)b * SEQ;

    // Q tile: 16 x 64 -> two A fragments (d 0..31 / 32..63), Q already pre-scaled
    FragAB qa[2];
#pragma unroll
    for (int f = 0; f < 2; f++)
#pragma unroll
        for (int j = 0; j < 8; j++) {
            int kk = f * 32 + kOffA(j, hi);
            qa[f].u[j] = *(const unsigned int*)(Qbase + (size_t)(q0 + ln) * DHEAD + kk);
        }

    unsigned qm = 0;                       // fg mask bits for the 16 query rows
    for (int i = 0; i < 16; i++) qm |= (mrow[q0 + i] ? 1u : 0u) << i;

    float zf[8], zb[8];
#pragma unroll
    for (int j = 0; j < 8; j++) { zf[j] = 0.f; zb[j] = 0.f; }

    // ---------------- pass 1: group partition functions ----------------
    for (int kc = 0; kc < SEQ; kc += 32) {
        const unsigned int* src = (const unsigned int*)(Kbase + (size_t)kc * DHEAD);
        for (int i = threadIdx.x; i < 1024; i += 128) {
            int k = i >> 5, dp = i & 31;
            ((unsigned int*)Kc)[k * 33 + dp] = src[i];
        }
        if (kc + 32 < SEQ)
            __builtin_prefetch((const char*)(src + 1024) + threadIdx.x * 32, 0, 0);
        __syncthreads();

#pragma unroll
        for (int t = 0; t < 2; t++) {
            f32x8 s = zero8();
#pragma unroll
            for (int f = 0; f < 2; f++) {
                FragAB kb;
#pragma unroll
                for (int j = 0; j < 8; j++) {
                    int d = f * 32 + hi * 16 + 2 * j;
                    kb.u[j] = *(const unsigned int*)(Kc + (t * 16 + ln) * 66 + d);
                }
                s = wmma_bf16(qa[f], kb, s);
            }
            const bool mk = mrow[kc + t * 16 + ln] != 0;
#pragma unroll
            for (int j = 0; j < 8; j++) {
                int row = j + 8 * hi;
                float e = __expf(s[j]);
                bool fg = mk && ((qm >> row) & 1);
                if (fg) zf[j] += e; else zb[j] += e;
            }
        }
        __syncthreads();
    }

    // reduce Z across the 16 lanes of each half (rows j+8*hi live in this half)
#pragma unroll
    for (int j = 0; j < 8; j++) {
#pragma unroll
        for (int off = 1; off < 16; off <<= 1) {
            zf[j] += __shfl_xor(zf[j], off, 32);
            zb[j] += __shfl_xor(zb[j], off, 32);
        }
    }
    float zfi[8], zbi[8];
#pragma unroll
    for (int j = 0; j < 8; j++) {
        zfi[j] = (zf[j] > 0.f) ? (1.f / zf[j]) : 0.f;
        zbi[j] = (zb[j] > 0.f) ? (1.f / zb[j]) : 0.f;
    }

    // ---------------- pass 2: probs + PV ----------------
    f32x8 oc[4] = {zero8(), zero8(), zero8(), zero8()};

    for (int kc = 0; kc < SEQ; kc += 32) {
        const unsigned int* ksrc = (const unsigned int*)(Kbase + (size_t)kc * DHEAD);
        const unsigned int* vsrc = (const unsigned int*)(Vbase + (size_t)kc * DHEAD);
        for (int i = threadIdx.x; i < 1024; i += 128) {
            int k = i >> 5, dp = i & 31;
            ((unsigned int*)Kc)[k * 33 + dp] = ksrc[i];
            // V staged transposed: Vt[d][k] so PV B-fragments are packed u32 reads
            unsigned int vv = vsrc[i];
            int d = dp * 2;
            Vt[d * 34 + k] = (unsigned short)vv;
            Vt[(d + 1) * 34 + k] = (unsigned short)(vv >> 16);
        }
        if (kc + 32 < SEQ) {
            __builtin_prefetch((const char*)(ksrc + 1024) + threadIdx.x * 32, 0, 0);
            __builtin_prefetch((const char*)(vsrc + 1024) + threadIdx.x * 32, 0, 0);
        }
        __syncthreads();

#pragma unroll
        for (int t = 0; t < 2; t++) {
            f32x8 s = zero8();
#pragma unroll
            for (int f = 0; f < 2; f++) {
                FragAB kb;
#pragma unroll
                for (int j = 0; j < 8; j++) {
                    int d = f * 32 + hi * 16 + 2 * j;
                    kb.u[j] = *(const unsigned int*)(Kc + (t * 16 + ln) * 66 + d);
                }
                s = wmma_bf16(qa[f], kb, s);
            }
            const bool mk = mrow[kc + t * 16 + ln] != 0;
#pragma unroll
            for (int j = 0; j < 8; j++) {
                int row = j + 8 * hi;
                bool fg = mk && ((qm >> row) & 1);
                float p = __expf(s[j]) * (fg ? zfi[j] : zbi[j]);
                Pb[w][row * 34 + t * 16 + ln] = f2bf(p);   // C-layout -> LDS
            }
        }

        // read P back in A-fragment layout (wave-internal LDS round trip)
        FragAB pa;
#pragma unroll
        for (int j = 0; j < 8; j++) {
            int kk = kOffA(j, hi);
            pa.u[j] = *(const unsigned int*)(&Pb[w][ln * 34 + kk]);
        }
#pragma unroll
        for (int dt = 0; dt < 4; dt++) {
            FragAB vb;
#pragma unroll
            for (int j = 0; j < 8; j++) {
                int kk = hi * 16 + 2 * j;
                vb.u[j] = *(const unsigned int*)(Vt + (dt * 16 + ln) * 34 + kk);
            }
            oc[dt] = wmma_bf16(pa, vb, oc[dt]);
        }
        __syncthreads();
    }

    // store ctx as bf16 in [B, S, H*Dh] (== transpose(0,2,1,3).reshape)
#pragma unroll
    for (int dt = 0; dt < 4; dt++)
#pragma unroll
        for (int j = 0; j < 8; j++) {
            int srow = q0 + j + 8 * hi;
            size_t idx = ((size_t)(b * SEQ + srow)) * DIM + h * DHEAD + dt * 16 + ln;
            ctxb[idx] = f2bf(oc[dt][j]);
        }
}

// ---------------- output projection GEMM (ctx @ Wo + bo -> f32 out) ----------------
__global__ void __launch_bounds__(128)
out_proj_kernel(const unsigned short* __restrict__ Xbf,
                const unsigned short* __restrict__ WT,
                const float* __restrict__ bo,
                float* __restrict__ out) {
    __shared__ __align__(16) unsigned short Bs[64 * 34];

    const int w = threadIdx.x >> 5;
    const int lane = threadIdx.x & 31;
    const int hi = lane >> 4, ln = lane & 15;

    const int bm = blockIdx.x / (DIM / 64);   // 0..31
    const int bn = blockIdx.x % (DIM / 64);   // 0..11
    const int m0 = bm * 128 + w * 32;
    const int n0 = bn * 64;

    f32x8 acc[2][4] = {{zero8(), zero8(), zero8(), zero8()},
                       {zero8(), zero8(), zero8(), zero8()}};

    for (int k0 = 0; k0 < DIM; k0 += 32) {
        __syncthreads();
        for (int i = threadIdx.x; i < 1024; i += 128) {
            int n = i >> 4, dp = i & 15;
            ((unsigned int*)Bs)[n * 17 + dp] =
                *(const unsigned int*)(WT + (size_t)(n0 + n) * DIM + k0 + dp * 2);
        }
        __syncthreads();

        FragAB a[2];
#pragma unroll
        for (int r = 0; r < 2; r++)
#pragma unroll
            for (int j = 0; j < 8; j++) {
                int kk = k0 + kOffA(j, hi);
                a[r].u[j] = *(const unsigned int*)(Xbf + (size_t)(m0 + r * 16 + ln) * DIM + kk);
            }
#pragma unroll
        for (int dt = 0; dt < 4; dt++) {
            FragAB bfr;
#pragma unroll
            for (int j = 0; j < 8; j++) {
                int kk = hi * 16 + 2 * j;
                bfr.u[j] = *(const unsigned int*)(Bs + (dt * 16 + ln) * 34 + kk);
            }
            acc[0][dt] = wmma_bf16(a[0], bfr, acc[0][dt]);
            acc[1][dt] = wmma_bf16(a[1], bfr, acc[1][dt]);
        }
    }

#pragma unroll
    for (int r = 0; r < 2; r++)
#pragma unroll
        for (int dt = 0; dt < 4; dt++)
#pragma unroll
            for (int j = 0; j < 8; j++) {
                int row = m0 + r * 16 + j + 8 * hi;
                int n = n0 + dt * 16 + ln;
                out[(size_t)row * DIM + n] = acc[r][dt][j] + bo[n];
            }
}

// ---------------- host-side launcher ----------------
extern "C" void kernel_launch(void* const* d_in, const int* in_sizes, int n_in,
                              void* d_out, int out_size, void* d_ws, size_t ws_size,
                              hipStream_t stream) {
    const float* hidden = (const float*)d_in[0];
    const float* Wq = (const float*)d_in[1];
    const float* bq = (const float*)d_in[2];
    const float* Wk = (const float*)d_in[3];
    const float* bk = (const float*)d_in[4];
    const float* Wv = (const float*)d_in[5];
    const float* bv = (const float*)d_in[6];
    const float* Wo = (const float*)d_in[7];
    const float* bo = (const float*)d_in[8];
    const unsigned char* mask = (const unsigned char*)d_in[9];
    float* out = (float*)d_out;

    char* w = (char*)d_ws;
    unsigned short* xb    = (unsigned short*)w; w += (size_t)MROWS * DIM * 2;        // 6.29 MB
    unsigned short* wtqkv = (unsigned short*)w; w += (size_t)NQKV * DIM * 2;         // 3.54 MB
    unsigned short* wto   = (unsigned short*)w; w += (size_t)DIM * DIM * 2;          // 1.18 MB
    float*          bqkv  = (float*)w;          w += (size_t)NQKV * 4;               // 9.2 KB
    unsigned short* Qh    = (unsigned short*)w; w += (size_t)BATCH * HEADS * SEQ * DHEAD * 2;
    unsigned short* Kh    = (unsigned short*)w; w += (size_t)BATCH * HEADS * SEQ * DHEAD * 2;
    unsigned short* Vh    = (unsigned short*)w; w += (size_t)BATCH * HEADS * SEQ * DHEAD * 2;
    unsigned short* ctxb  = (unsigned short*)w; w += (size_t)MROWS * DIM * 2;        // total ~37 MB

    pack_hidden_kernel<<<3072, 256, 0, stream>>>(hidden, xb, MROWS * DIM);
    pack_weights_kernel<<<((NQKV + DIM) * DIM + 255) / 256, 256, 0, stream>>>(
        Wq, Wk, Wv, Wo, wtqkv, wto);
    pack_bias_kernel<<<(NQKV + 255) / 256, 256, 0, stream>>>(bq, bk, bv, bqkv);

    // block tile 128x64, 4 waves (32x64 each); grid 32 x 36
    qkv_gemm_kernel<<<32 * (NQKV / 64), 128, 0, stream>>>(xb, wtqkv, bqkv, Qh, Kh, Vh);

    // B*H*32 blocks, 4 waves/block (64 queries per block)
    attn_kernel<<<BATCH * HEADS * 32, 128, 0, stream>>>(Qh, Kh, Vh, mask, ctxb);

    // block tile 128x64; grid 32 x 12
    out_proj_kernel<<<32 * (DIM / 64), 128, 0, stream>>>(ctxb, wto, bo, out);
}
